// PotentialNetParallel_16174846837228
// MI455X (gfx1250) — compile-verified, matrix operand order checked by reading, not run on hardware
//
#include <hip/hip_runtime.h>

#define N_NODES 100000
#define N_EDGES 1000000
#define N_GRAPHS 64
#define IN_CH 19
#define GW1 128
#define GW2 64

typedef __attribute__((ext_vector_type(16))) __bf16 v16bf;
typedef __attribute__((ext_vector_type(8)))  float  v8f;

// fast softsign: denominator >= 1, so single-instruction v_rcp_f32 is safe
__device__ __forceinline__ float ss(float z) {
    return z * __builtin_amdgcn_rcpf(1.0f + fabsf(z));
}

// K index inside a 32-wide K-step for 16-bit A/B operands (wave32 layout):
// lane<16 holds K {0..7, 16..23}; lane>=16 holds K {8..15, 24..31}; row/col = lane%16
__device__ __forceinline__ int a_kidx(int lane, int i) {
    int lo = (lane & 16) ? 8 : 0;
    return (i < 8) ? (lo + i) : (lo + 16 + (i - 8));
}

__device__ __forceinline__ v8f wmma_bf16(v16bf a, v16bf b, v8f c) {
    return __builtin_amdgcn_wmma_f32_16x16x32_bf16(
        false, a, false, b, (short)0, c, false, false);
}

// Build A operand from an f32 row (128 floats), K-step ks
__device__ __forceinline__ v16bf a_from_f32row(const float* row, int ks, int lane) {
    int lo = (lane & 16) ? 8 : 0;
    const float* r = row + ks * 32 + lo;
    v16bf t;
#pragma unroll
    for (int i = 0; i < 8; i++) t[i] = (__bf16)r[i];
#pragma unroll
    for (int i = 0; i < 8; i++) t[i + 8] = (__bf16)r[16 + i];
    return t;
}

// Build A operand from a bf16 row in LDS (128 __bf16), K-step ks
__device__ __forceinline__ v16bf a_from_bf16row(const __bf16* row, int ks, int lane) {
    int lo = (lane & 16) ? 8 : 0;
    const __bf16* r = row + ks * 32 + lo;
    v16bf t;
#pragma unroll
    for (int i = 0; i < 8; i++) t[i] = r[i];
#pragma unroll
    for (int i = 0; i < 8; i++) t[i + 8] = r[16 + i];
    return t;
}

// ---------------- utility kernels ----------------

__global__ void k_zero(float* p, long n) {
    long i = (long)blockIdx.x * blockDim.x + threadIdx.x;
    if (i < n) p[i] = 0.0f;
}

// Pre-swizzle a [K x Nc] f32 weight matrix into per-lane-contiguous bf16 B-operand
// tiles: out[((nt*KS + ks)*32 + lane)*16 + i] = bf16(W[(ks*32 + kidx(lane,i))*Nc + nt*16 + lane%16])
__global__ void k_swzB(const float* W, __bf16* out, int K, int Nc) {
    int KS = K / 32;
    int total = (Nc / 16) * KS * 32 * 16;
    int idx = blockIdx.x * blockDim.x + threadIdx.x;
    if (idx >= total) return;
    int i    = idx & 15;
    int lane = (idx >> 4) & 31;
    int rest = idx >> 9;
    int ks   = rest % KS;
    int nt   = rest / KS;
    out[idx] = (__bf16)W[(ks * 32 + a_kidx(lane, i)) * Nc + nt * 16 + (lane & 15)];
}

// ---------------- stage 1 (cov, feat=19): scalar f32 ----------------

__global__ __launch_bounds__(256) void k_edge1(
    const float* __restrict__ x, const int* __restrict__ ei, const float* __restrict__ ea,
    const float* eW1, const float* eb1, const float* eW2, const float* eb2,
    float* agg1) {
    int e = blockIdx.x * blockDim.x + threadIdx.x;
    if (e >= N_EDGES) return;
    float a = ea[e];
    if (a > 0.3f) return;                       // cov threshold mask
    int src = ei[e], dst = ei[N_EDGES + e];
    float hm[9];
#pragma unroll
    for (int j = 0; j < 9; j++) hm[j] = ss(a * eW1[j] + eb1[j]);
    const float* xs = x + (long)src * IN_CH;
    float* ag = agg1 + (long)dst * IN_CH;
#pragma unroll
    for (int c = 0; c < IN_CH; c++) {
        float z = eb2[c];
#pragma unroll
        for (int j = 0; j < 9; j++) z += hm[j] * eW2[j * IN_CH + c];
        atomicAdd(&ag[c], xs[c] * ss(z));
    }
}

__global__ __launch_bounds__(64) void k_node1(
    const float* __restrict__ x, const float* __restrict__ agg1,
    const float* Wr, const float* br,
    const float* Wi1, const float* bi1, const float* Wi2, const float* bi2,
    const float* Wj, const float* bj,
    float* __restrict__ h1) {
    __shared__ float logit[64][GW1];
    int n = blockIdx.x * 64 + threadIdx.x;
    if (n >= N_NODES) return;
    float xv[IN_CH];
#pragma unroll
    for (int k = 0; k < IN_CH; k++) xv[k] = x[(long)n * IN_CH + k];
    float h1p[IN_CH];
    for (int c = 0; c < IN_CH; c++) {
        float z = agg1[(long)n * IN_CH + c] + br[c];
        for (int k = 0; k < IN_CH; k++) z += xv[k] * Wr[k * IN_CH + c];
        h1p[c] = z;
    }
    float zi[IN_CH];
    for (int c = 0; c < IN_CH; c++) {
        float z = bi1[c];
        for (int k = 0; k < IN_CH; k++) z += h1p[k] * Wi1[k * IN_CH + c];
        for (int k = 0; k < IN_CH; k++) z += xv[k] * Wi1[(IN_CH + k) * IN_CH + c];
        zi[c] = ss(z);
    }
    float* lrow = logit[threadIdx.x];
    float mx = -1e30f;
    for (int c = 0; c < GW1; c++) {
        float z = bi2[c];
        for (int k = 0; k < IN_CH; k++) z += zi[k] * Wi2[k * GW1 + c];
        z = ss(z);
        lrow[c] = z;
        mx = fmaxf(mx, z);
    }
    float s = 0.0f;
    for (int c = 0; c < GW1; c++) { float e = __expf(lrow[c] - mx); lrow[c] = e; s += e; }
    float inv = __builtin_amdgcn_rcpf(s);
    for (int c = 0; c < GW1; c++) {
        float z = bj[c];
        for (int k = 0; k < IN_CH; k++) z += xv[k] * Wj[k * GW1 + c];
        h1[(long)n * GW1 + c] = lrow[c] * inv * ss(z);
    }
}

// ---------------- stage 2 (noncov) edge kernel: WMMA 16 edges/wave ----------------

__global__ __launch_bounds__(128) void k_edge2(
    const float* __restrict__ h1, const int* __restrict__ ei, const float* __restrict__ ea,
    const float* eW1, const float* eb1,
    const __bf16* __restrict__ W2swz, const float* eb2,
    float* agg2) {
    int lane = threadIdx.x & 31;
    int wid  = threadIdx.x >> 5;
    int tile = blockIdx.x * 4 + wid;
    int e0 = tile * 16;
    if (e0 >= N_EDGES) return;

    // A: hidden = softsign(ea*W1 + b1) computed straight into the A-operand layout
    float av = ea[e0 + (lane & 15)];
    int lo = (lane & 16) ? 8 : 0;
    v16bf a[2];
#pragma unroll
    for (int ks = 0; ks < 2; ks++) {
#pragma unroll
        for (int i = 0; i < 16; i++) {
            int kk = ks * 32 + ((i < 8) ? (lo + i) : (lo + 16 + i - 8));
            a[ks][i] = (__bf16)ss(av * eW1[kk] + eb1[kk]);
        }
    }

    int   srcv[8], dstv[8];
    float eav[8];
    int mbase = (lane & 16) ? 8 : 0;
#pragma unroll
    for (int r = 0; r < 8; r++) {
        int e = e0 + mbase + r;
        srcv[r] = ei[e];
        dstv[r] = ei[N_EDGES + e];
        eav[r]  = ea[e];
        if (eav[r] <= 0.7f)
            __builtin_prefetch(h1 + (long)srcv[r] * GW1 + (lane & 15), 0, 0);
    }

#pragma unroll 1
    for (int nt = 0; nt < 8; nt++) {
        v8f c = {};
#pragma unroll
        for (int ks = 0; ks < 2; ks++) {
            v16bf b = *(const v16bf*)(W2swz + (((nt * 2 + ks) * 32 + lane) << 4));
            c = wmma_bf16(a[ks], b, c);
        }
        int col = nt * 16 + (lane & 15);
        float b2c = eb2[col];
#pragma unroll
        for (int r = 0; r < 8; r++) {
            if (eav[r] <= 0.7f) {                       // noncov threshold mask
                float w = ss(c[r] + b2c);
                float msg = h1[(long)srcv[r] * GW1 + col] * w;
                atomicAdd(&agg2[(long)dstv[r] * GW1 + col], msg);
            }
        }
    }
}

// ---------------- stage 2 (noncov) node kernel: WMMA 16 nodes/wave ----------------

__global__ __launch_bounds__(128) void k_node2(
    const float* __restrict__ x, const float* __restrict__ h1, const float* __restrict__ agg2,
    const __bf16* __restrict__ WrS,  const float* br,
    const __bf16* __restrict__ Wi1S, const float* bi1,
    const __bf16* __restrict__ Wi2S, const float* bi2,
    const __bf16* __restrict__ WjS,  const float* bj,
    float* __restrict__ h2) {
    __shared__ __bf16 sPre[4][16 * GW1];   // h1pre tile, bf16
    __shared__ __bf16 sZi [4][16 * GW1];   // zi tile, bf16
    __shared__ float  sAtt[4][16 * GW2];   // attention logits/weights

    int lane = threadIdx.x & 31;
    int wid  = threadIdx.x >> 5;
    const int NTILES = N_NODES / 16;
    int tile = blockIdx.x * 4 + wid;
    if (tile >= NTILES) tile = NTILES - 1;         // clamp: duplicate recompute, uniform barriers
    int n0 = tile * 16;
    int mrow  = lane & 15;
    int mbase = (lane & 16) ? 8 : 0;

    __bf16* pre  = sPre[wid];
    __bf16* zib  = sZi[wid];
    float*  attb = sAtt[wid];

    // cache h1 A-operands (reused by root matmul, net_i tail-K, net_j)
    const float* arow = h1 + (long)(n0 + mrow) * GW1;
    v16bf aG[4];
#pragma unroll
    for (int ks = 0; ks < 4; ks++) aG[ks] = a_from_f32row(arow, ks, lane);

    // Phase 1: h1pre = agg2 + h1 @ Wr + br    (8 nt x 4 ks = 32 WMMA)
#pragma unroll 1
    for (int nt = 0; nt < 8; nt++) {
        v8f c = {};
#pragma unroll
        for (int ks = 0; ks < 4; ks++) {
            v16bf b = *(const v16bf*)(WrS + (((nt * 4 + ks) * 32 + lane) << 4));
            c = wmma_bf16(aG[ks], b, c);
        }
        int col = nt * 16 + (lane & 15);
#pragma unroll
        for (int r = 0; r < 8; r++) {
            int M = mbase + r;
            float v = c[r] + agg2[(long)(n0 + M) * GW1 + col] + br[col];
            pre[M * GW1 + col] = (__bf16)v;
        }
    }
    __syncthreads();

    // Phase 2: zi = softsign([h1pre, h1] @ Wi1 + bi1), K=256  (8 nt x 8 ks = 64 WMMA)
    const __bf16* prow = pre + mrow * GW1;
#pragma unroll 1
    for (int nt = 0; nt < 8; nt++) {
        v8f c = {};
#pragma unroll
        for (int ks = 0; ks < 8; ks++) {
            v16bf aa = (ks < 4) ? a_from_bf16row(prow, ks, lane) : aG[ks - 4];
            v16bf b  = *(const v16bf*)(Wi1S + (((nt * 8 + ks) * 32 + lane) << 4));
            c = wmma_bf16(aa, b, c);
        }
        int col = nt * 16 + (lane & 15);
#pragma unroll
        for (int r = 0; r < 8; r++) {
            int M = mbase + r;
            zib[M * GW1 + col] = (__bf16)ss(c[r] + bi1[col]);
        }
    }
    __syncthreads();

    // Phase 3: attention logits = softsign(zi @ Wi2 + bi2)  (4 nt x 4 ks = 16 WMMA)
    const __bf16* zrow = zib + mrow * GW1;
#pragma unroll 1
    for (int nt = 0; nt < 4; nt++) {
        v8f c = {};
#pragma unroll
        for (int ks = 0; ks < 4; ks++) {
            v16bf aa = a_from_bf16row(zrow, ks, lane);
            v16bf b  = *(const v16bf*)(Wi2S + (((nt * 4 + ks) * 32 + lane) << 4));
            c = wmma_bf16(aa, b, c);
        }
        int col = nt * 16 + (lane & 15);
#pragma unroll
        for (int r = 0; r < 8; r++) {
            int M = mbase + r;
            attb[M * GW2 + col] = ss(c[r] + bi2[col]);
        }
    }
    __syncthreads();

    // Phase 4: row softmax over 64 features (lanes 0..15 handle one row each)
    if (lane < 16) {
        float* rp = attb + lane * GW2;
        float mx = -1e30f;
        for (int c = 0; c < GW2; c++) mx = fmaxf(mx, rp[c]);
        float s = 0.0f;
        for (int c = 0; c < GW2; c++) { float e = __expf(rp[c] - mx); rp[c] = e; s += e; }
        float inv = __builtin_amdgcn_rcpf(s);
        for (int c = 0; c < GW2; c++) rp[c] *= inv;
    }
    __syncthreads();

    // Phase 5: h2 = att * softsign(h1 @ Wj + bj), ligand mask  (16 WMMA)
#pragma unroll 1
    for (int nt = 0; nt < 4; nt++) {
        v8f c = {};
#pragma unroll
        for (int ks = 0; ks < 4; ks++) {
            v16bf b = *(const v16bf*)(WjS + (((nt * 4 + ks) * 32 + lane) << 4));
            c = wmma_bf16(aG[ks], b, c);
        }
        int col = nt * 16 + (lane & 15);
#pragma unroll
        for (int r = 0; r < 8; r++) {
            int M = mbase + r;
            int n = n0 + M;
            float z = ss(c[r] + bj[col]);
            float flag = x[(long)n * IN_CH + 14];
            float val = (flag == -1.0f) ? 0.0f : attb[M * GW2 + col] * z;
            h2[(long)n * GW2 + col] = val;
        }
    }
}

// ---------------- pooling + FC head ----------------

__global__ __launch_bounds__(256) void k_pool(
    const float* __restrict__ h2, const int* __restrict__ batch, float* pool) {
    __shared__ float part[N_GRAPHS * GW2];       // 16 KB per-block partials
    for (int i = threadIdx.x; i < N_GRAPHS * GW2; i += 256) part[i] = 0.0f;
    __syncthreads();
    int n0 = blockIdx.x * 512;
    int n1 = n0 + 512; if (n1 > N_NODES) n1 = N_NODES;
    for (long idx = (long)n0 * GW2 + threadIdx.x; idx < (long)n1 * GW2; idx += 256) {
        int n = (int)(idx / GW2), c = (int)(idx % GW2);
        atomicAdd(&part[batch[n] * GW2 + c], h2[idx]);
    }
    __syncthreads();
    for (int i = threadIdx.x; i < N_GRAPHS * GW2; i += 256)
        if (part[i] != 0.0f) atomicAdd(&pool[i], part[i]);
}

__global__ void k_fc(const float* __restrict__ pool,
                     const float* W1, const float* b1,
                     const float* W2, const float* b2,
                     const float* W3, const float* b3,
                     float* out) {
    int g = threadIdx.x;
    if (g >= N_GRAPHS) return;
    const float* pr = pool + g * GW2;
    float z1[42];
    for (int j = 0; j < 42; j++) {
        float z = b1[j];
        for (int k = 0; k < 64; k++) z += pr[k] * W1[k * 42 + j];
        z1[j] = fmaxf(z, 0.0f);
    }
    float z2[32];
    for (int j = 0; j < 32; j++) {
        float z = b2[j];
        for (int k = 0; k < 42; k++) z += z1[k] * W2[k * 32 + j];
        z2[j] = fmaxf(z, 0.0f);
    }
    float z = b3[0];
    for (int k = 0; k < 32; k++) z += z2[k] * W3[k];
    out[g] = z;
}

// ---------------- launcher ----------------

extern "C" void kernel_launch(void* const* d_in, const int* in_sizes, int n_in,
                              void* d_out, int out_size, void* d_ws, size_t ws_size,
                              hipStream_t stream) {
    (void)in_sizes; (void)n_in; (void)out_size; (void)ws_size;
    const float* x     = (const float*)d_in[0];
    const int*   ei    = (const int*)  d_in[1];
    const float* ea    = (const float*)d_in[2];
    const int*   batch = (const int*)  d_in[3];
    // cov params
    const float* c_eW1 = (const float*)d_in[4];
    const float* c_eb1 = (const float*)d_in[5];
    const float* c_eW2 = (const float*)d_in[6];
    const float* c_eb2 = (const float*)d_in[7];
    const float* c_Wr  = (const float*)d_in[8];
    const float* c_br  = (const float*)d_in[9];
    const float* c_Wi1 = (const float*)d_in[10];
    const float* c_bi1 = (const float*)d_in[11];
    const float* c_Wi2 = (const float*)d_in[12];
    const float* c_bi2 = (const float*)d_in[13];
    const float* c_Wj  = (const float*)d_in[14];
    const float* c_bj  = (const float*)d_in[15];
    // noncov params
    const float* n_eW1 = (const float*)d_in[16];
    const float* n_eb1 = (const float*)d_in[17];
    const float* n_eW2 = (const float*)d_in[18];
    const float* n_eb2 = (const float*)d_in[19];
    const float* n_Wr  = (const float*)d_in[20];
    const float* n_br  = (const float*)d_in[21];
    const float* n_Wi1 = (const float*)d_in[22];
    const float* n_bi1 = (const float*)d_in[23];
    const float* n_Wi2 = (const float*)d_in[24];
    const float* n_bi2 = (const float*)d_in[25];
    const float* n_Wj  = (const float*)d_in[26];
    const float* n_bj  = (const float*)d_in[27];
    // fc params
    const float* fW1 = (const float*)d_in[28];
    const float* fb1 = (const float*)d_in[29];
    const float* fW2 = (const float*)d_in[30];
    const float* fb2 = (const float*)d_in[31];
    const float* fW3 = (const float*)d_in[32];
    const float* fb3 = (const float*)d_in[33];

    char* p = (char*)d_ws;
    auto alloc = [&](size_t bytes) { char* r = p; p += (bytes + 255) & ~(size_t)255; return (void*)r; };
    float* agg1 = (float*)alloc(sizeof(float) * (size_t)N_NODES * IN_CH);
    float* h1   = (float*)alloc(sizeof(float) * (size_t)N_NODES * GW1);
    float* agg2 = (float*)alloc(sizeof(float) * (size_t)N_NODES * GW1);
    float* h2   = (float*)alloc(sizeof(float) * (size_t)N_NODES * GW2);
    float* pool = (float*)alloc(sizeof(float) * N_GRAPHS * GW2);
    __bf16* W2eS = (__bf16*)alloc(2u * 64 * 128);
    __bf16* WrS  = (__bf16*)alloc(2u * 128 * 128);
    __bf16* Wi1S = (__bf16*)alloc(2u * 256 * 128);
    __bf16* Wi2S = (__bf16*)alloc(2u * 128 * 64);
    __bf16* WjS  = (__bf16*)alloc(2u * 128 * 64);

    // zero accumulators (must be per-call: atomics accumulate)
    long nAgg1 = (long)N_NODES * IN_CH, nAgg2 = (long)N_NODES * GW1, nPool = N_GRAPHS * GW2;
    k_zero<<<(unsigned)((nAgg1 + 255) / 256), 256, 0, stream>>>(agg1, nAgg1);
    k_zero<<<(unsigned)((nAgg2 + 255) / 256), 256, 0, stream>>>(agg2, nAgg2);
    k_zero<<<(unsigned)((nPool + 255) / 256), 256, 0, stream>>>(pool, nPool);

    // swizzle noncov weights into WMMA B-operand layout (bf16)
    k_swzB<<<(64 * 128 + 255) / 256, 256, 0, stream>>>(n_eW2, W2eS, 64, 128);
    k_swzB<<<(128 * 128 + 255) / 256, 256, 0, stream>>>(n_Wr, WrS, 128, 128);
    k_swzB<<<(256 * 128 + 255) / 256, 256, 0, stream>>>(n_Wi1, Wi1S, 256, 128);
    k_swzB<<<(128 * 64 + 255) / 256, 256, 0, stream>>>(n_Wi2, Wi2S, 128, 64);
    k_swzB<<<(128 * 64 + 255) / 256, 256, 0, stream>>>(n_Wj, WjS, 128, 64);

    // stage 1 (cov, feat=19): scalar f32
    k_edge1<<<(N_EDGES + 255) / 256, 256, 0, stream>>>(x, ei, ea, c_eW1, c_eb1, c_eW2, c_eb2, agg1);
    k_node1<<<(N_NODES + 63) / 64, 64, 0, stream>>>(x, agg1, c_Wr, c_br,
                                                    c_Wi1, c_bi1, c_Wi2, c_bi2, c_Wj, c_bj, h1);

    // stage 2 (noncov, feat=128): WMMA bf16
    k_edge2<<<(N_EDGES / 16) / 4, 128, 0, stream>>>(h1, ei, ea, n_eW1, n_eb1, W2eS, n_eb2, agg2);
    k_node2<<<(N_NODES / 16 + 3) / 4, 128, 0, stream>>>(x, h1, agg2,
                                                        WrS, n_br, Wi1S, n_bi1,
                                                        Wi2S, n_bi2, WjS, n_bj, h2);

    // pool + FC head
    k_pool<<<(N_NODES + 511) / 512, 256, 0, stream>>>(h2, batch, pool);
    k_fc<<<1, 64, 0, stream>>>(pool, fW1, fb1, fW2, fb2, fW3, fb3, (float*)d_out);
}
